// StackGCNEncoder_5428838662652
// MI455X (gfx1250) — matrix operand order repaired and checked by reading, not compiled, not run on hardware
//
#include <hip/hip_runtime.h>
#include <stdint.h>

// ---------------- problem constants ----------------
#define S      5
#define NNODE  2000      // NL == ND
#define DIN    4000
#define DSUB   250
#define NNZ    100000
#define ALPHA  50.0f
#define BETA   1.0f

#define NPAD   256            // DSUB padded to multiple of 16
#define NT     (NPAD / 16)    // 16 N-tiles
#define NTW    4              // N-tiles per wave (register blocking)
#define KT1    (DIN / 32)     // 125 K-tiles for projection GEMM
#define KPAD   2016           // NNODE padded to multiple of 32
#define KT2    (KPAD / 32)    // 63 K-tiles for attention GEMM
#define MTILES (NNODE / 16)   // 125 M-tiles

typedef __attribute__((ext_vector_type(16))) __bf16 v16bf;
typedef __attribute__((ext_vector_type(8)))  __bf16 v8bf;
typedef __attribute__((ext_vector_type(8)))  float  v8f;

// Build a 16-element bf16 fragment from two contiguous 8-element runs.
__device__ __forceinline__ v16bf frag_from(const __bf16* pa, const __bf16* pb) {
  v8bf x = *(const v8bf*)pa;
  v8bf y = *(const v8bf*)pb;
  v16bf r;
#pragma unroll
  for (int i = 0; i < 8; ++i) { r[i] = x[i]; r[8 + i] = y[i]; }
  return r;
}

// ---------------- elementwise helpers ----------------
__global__ void cvt_f32_bf16(const float* __restrict__ src, __bf16* __restrict__ dst, int n) {
  int i = blockIdx.x * blockDim.x + threadIdx.x;
  if (i < n) dst[i] = (__bf16)src[i];
}

__global__ void zero_f32(float* __restrict__ p, int n) {
  int i = blockIdx.x * blockDim.x + threadIdx.x;
  if (i < n) p[i] = 0.0f;
}

// ---------------- B-matrix packing ----------------
// B packing (this code's k order, same as A) for v_wmma_f32_16x16x32_bf16 (32x16 KxN): lane = n + 16*hi,
// per-lane 16 contiguous K values: k = hi*16 + idx.
// Packed index: (((nb)*KT + kb)*512) + lane*16 + idx  (per matrix).
__global__ void pack_w(const float* __restrict__ W, __bf16* __restrict__ Bp) {
  int t = blockIdx.x * blockDim.x + threadIdx.x;
  const int total = S * NT * KT1 * 512;
  if (t >= total) return;
  int e = t & 511;
  int tile = t >> 9;
  int kb = tile % KT1; tile /= KT1;
  int nb = tile % NT;
  int s  = tile / NT;
  int lane = e >> 4, idx = e & 15;
  int k = kb * 32 + (lane >> 4) * 16 + idx;
  int n = nb * 16 + (lane & 15);
  float v = (n < DSUB) ? W[(size_t)s * DIN * DSUB + (size_t)k * DSUB + n] : 0.0f;
  Bp[t] = (__bf16)v;
}

__global__ void pack_h(const float* __restrict__ H, __bf16* __restrict__ Bp) {
  int t = blockIdx.x * blockDim.x + threadIdx.x;
  const int total = NT * KT2 * 512;
  if (t >= total) return;
  int e = t & 511;
  int tile = t >> 9;
  int kb = tile % KT2;
  int nb = tile / KT2;
  int lane = e >> 4, idx = e & 15;
  int k = kb * 32 + (lane >> 4) * 16 + idx;
  int n = nb * 16 + (lane & 15);
  float v = (k < NNODE) ? H[(size_t)k * NPAD + n] : 0.0f;   // NPAD cols, pad cols are 0
  Bp[t] = (__bf16)v;
}

// ---------------- WMMA GEMM: C[M,NPAD] = A_bf16[M,K] @ Bpacked ----------------
// One wave per 16x(16*NTW) output tile: A fragment reused NTW times.
// KT is a template constant so all 4 B streams index off ONE base pointer with
// immediate offsets -> global_load_b128 (not flat), no per-step pointer math.
template <int KT>
__global__ void gemm_bf16(const __bf16* __restrict__ A, int lda,
                          const __bf16* __restrict__ Bp,
                          float* __restrict__ C, int ldc) {
  int lane = threadIdx.x;            // 0..31
  int hi = lane >> 4;
  int mt = blockIdx.x;
  int nt0 = blockIdx.y * NTW;
  const __bf16* arow = A + (size_t)(mt * 16 + (lane & 15)) * lda;
  const __bf16* bbase = Bp + (size_t)nt0 * KT * 512 + lane * 16;
  v8f acc[NTW];
#pragma unroll
  for (int j = 0; j < NTW; ++j) acc[j] = (v8f){};

  for (int kb = 0; kb < KT; ++kb) {
    int k0 = kb * 32 + hi * 16;      // A k order used here (same as B packing): contiguous run [k0,k0+16)
    __builtin_prefetch(arow + k0 + 4 * 32, 0, 3);   // stream A ahead (global_prefetch)
    v16bf a = frag_from(arow + k0, arow + k0 + 8);
#pragma unroll
    for (int j = 0; j < NTW; ++j) {
      const __bf16* bp = bbase + ((size_t)j * KT + kb) * 512;
      v16bf b = frag_from(bp, bp + 8);
      acc[j] = __builtin_amdgcn_wmma_f32_16x16x32_bf16(false, a, false, b,
                                                       (short)0, acc[j], false, false);
    }
  }
  int rowbase = mt * 16 + hi * 8;
#pragma unroll
  for (int j = 0; j < NTW; ++j) {
    int col = (nt0 + j) * 16 + (lane & 15);
#pragma unroll
    for (int g = 0; g < 8; ++g)
      C[(size_t)(rowbase + g) * ldc + col] = acc[j][g];
  }
}

// ---------------- COO SpMM: Y[s,r,:] += v * X[s,c,:] ----------------
__global__ void spmm_atomic(const int* __restrict__ rows, const int* __restrict__ cols,
                            const float* __restrict__ vals, const float* __restrict__ X,
                            float* __restrict__ Y) {
  int e = blockIdx.x, s = blockIdx.y, d = threadIdx.x;
  if (d >= DSUB) return;
  size_t o = (size_t)s * NNZ + e;
  int r = rows[o], c = cols[o];
  float v = vals[o];
  atomicAdd(&Y[((size_t)s * NNODE + r) * NPAD + d],
            v * X[((size_t)s * NNODE + c) * NPAD + d]);
}

// ---------------- fold conv1x1 + f vectors: g = f_w @ caw, c = dot(cab,f_w)+f_b --------
__global__ void gvec(const float* __restrict__ caw, const float* __restrict__ cab,
                     const float* __restrict__ f1w, const float* __restrict__ f1b,
                     const float* __restrict__ f2w, const float* __restrict__ f2b,
                     float* __restrict__ g1, float* __restrict__ g2,
                     float* __restrict__ cc) {
  int k = threadIdx.x;
  if (k < DSUB) {
    float s1 = 0.f, s2 = 0.f;
    for (int j = 0; j < DSUB; ++j) {
      float c = caw[j * DSUB + k];
      s1 += f1w[j] * c;
      s2 += f2w[j] * c;
    }
    g1[k] = s1; g2[k] = s2;
  }
  if (k == 0) {
    float c1 = f1b[0], c2 = f2b[0];
    for (int j = 0; j < DSUB; ++j) { c1 += cab[j] * f1w[j]; c2 += cab[j] * f2w[j]; }
    cc[0] = c1; cc[1] = c2;
  }
}

// ---------------- f1/f2 per node: one wave per node, coalesced + shuffle reduce --------
__global__ void fvals(const float* __restrict__ H, const float* __restrict__ g1,
                      const float* __restrict__ g2, const float* __restrict__ cc,
                      float* __restrict__ F1, float* __restrict__ F2) {
  int n = blockIdx.x, s = blockIdx.y, lane = threadIdx.x;   // 32 threads
  const float* h = H + ((size_t)s * NNODE + n) * NPAD;
  float a = 0.f, b = 0.f;
  for (int d = lane; d < DSUB; d += 32) {
    float x = h[d];
    a += x * g1[d];
    b += x * g2[d];
  }
#pragma unroll
  for (int sh = 16; sh > 0; sh >>= 1) {
    a += __shfl_down(a, sh, 32);
    b += __shfl_down(b, sh, 32);
  }
  if (lane == 0) {
    F1[s * NNODE + n] = a + cc[0];
    F2[s * NNODE + n] = b + cc[1];
  }
}

// ---------------- row softmax -> unnormalized exp (bf16) + row sum Z --------------
__global__ void softmax_rows(const float* __restrict__ F1, const float* __restrict__ F2,
                             const float* __restrict__ Wside, __bf16* __restrict__ P,
                             float* __restrict__ Z) {
  __shared__ float red[256];
  int n = blockIdx.x, t = threadIdx.x;
  float f1 = F1[n];
  const float* wr = Wside + (size_t)n * NNODE;
  float mx = -1e30f;
  for (int m = t; m < NNODE; m += 256) {
    float l = f1 + F2[m];
    l = (l > 0.f) ? l : 0.1f * l;       // leaky_relu(.,0.1)
    l += wr[m];
    mx = fmaxf(mx, l);
  }
  red[t] = mx; __syncthreads();
  for (int s = 128; s > 0; s >>= 1) { if (t < s) red[t] = fmaxf(red[t], red[t + s]); __syncthreads(); }
  mx = red[0]; __syncthreads();
  float sum = 0.f;
  for (int m = t; m < NNODE; m += 256) {
    float l = f1 + F2[m];
    l = (l > 0.f) ? l : 0.1f * l;
    l += wr[m];
    float p = __expf(l - mx);
    sum += p;
    P[(size_t)n * KPAD + m] = (__bf16)p;
  }
  for (int m = NNODE + t; m < KPAD; m += 256) P[(size_t)n * KPAD + m] = (__bf16)0.f;
  red[t] = sum; __syncthreads();
  for (int s = 128; s > 0; s >>= 1) { if (t < s) red[t] += red[t + s]; __syncthreads(); }
  if (t == 0) Z[n] = red[0];
}

// ------- attention GEMM (P @ hidden) with fused CRF epilogue + ReLU + concat store ------
__global__ void attn_gemm(const __bf16* __restrict__ P, const __bf16* __restrict__ Bp,
                          const float* __restrict__ Z, const float* __restrict__ H,
                          float* __restrict__ Out, int sIdx) {
  int lane = threadIdx.x;
  int hi = lane >> 4;
  int mt = blockIdx.x;
  int nt0 = blockIdx.y * NTW;
  const __bf16* arow = P + (size_t)(mt * 16 + (lane & 15)) * KPAD;
  const __bf16* bbase = Bp + (size_t)nt0 * KT2 * 512 + lane * 16;
  v8f acc[NTW];
#pragma unroll
  for (int j = 0; j < NTW; ++j) acc[j] = (v8f){};

  for (int kb = 0; kb < KT2; ++kb) {
    int k0 = kb * 32 + hi * 16;
    __builtin_prefetch(arow + k0 + 4 * 32, 0, 3);
    v16bf a = frag_from(arow + k0, arow + k0 + 8);
#pragma unroll
    for (int j = 0; j < NTW; ++j) {
      const __bf16* bp = bbase + ((size_t)j * KT2 + kb) * 512;
      v16bf b = frag_from(bp, bp + 8);
      acc[j] = __builtin_amdgcn_wmma_f32_16x16x32_bf16(false, a, false, b,
                                                       (short)0, acc[j], false, false);
    }
  }
  int rowbase = mt * 16 + hi * 8;
#pragma unroll
  for (int j = 0; j < NTW; ++j) {
    int col = (nt0 + j) * 16 + (lane & 15);
    if (col < DSUB) {
#pragma unroll
      for (int g = 0; g < 8; ++g) {
        int row = rowbase + g;
        float z = Z[row];
        float hv = H[(size_t)row * NPAD + col];
        float y = (BETA * acc[j][g] / z + ALPHA * hv) / (BETA + ALPHA);   // rowsum == 1
        Out[(size_t)row * (S * DSUB) + (size_t)sIdx * DSUB + col] = fmaxf(y, 0.f);
      }
    }
  }
}

// ---------------- launch ----------------
extern "C" void kernel_launch(void* const* d_in, const int* in_sizes, int n_in,
                              void* d_out, int out_size, void* d_ws, size_t ws_size,
                              hipStream_t stream) {
  (void)in_sizes; (void)n_in; (void)out_size; (void)ws_size;

  const float* lnc_in   = (const float*)d_in[0];
  const float* dis_in   = (const float*)d_in[1];
  const float* lnc_vals = (const float*)d_in[2];
  const float* dis_vals = (const float*)d_in[3];
  const float* weight   = (const float*)d_in[4];
  const float* w_lnc    = (const float*)d_in[5];   // [1,N,N]
  const float* w_dis    = (const float*)d_in[6];
  const float* caw      = (const float*)d_in[7];
  const float* cab      = (const float*)d_in[8];
  const float* f1w      = (const float*)d_in[9];
  const float* f1b      = (const float*)d_in[10];
  const float* f2w      = (const float*)d_in[11];
  const float* f2b      = (const float*)d_in[12];
  const int* lnc_rows   = (const int*)d_in[13];
  const int* lnc_cols   = (const int*)d_in[14];
  const int* dis_rows   = (const int*)d_in[15];
  const int* dis_cols   = (const int*)d_in[16];
  float* out = (float*)d_out;

  // ---- workspace carve-up (256B aligned) ----
  char* ws = (char*)d_ws;
  size_t off = 0;
  auto alloc = [&](size_t bytes) { size_t o = off; off += (bytes + 255) & ~(size_t)255; return o; };

  __bf16* lncB = (__bf16*)(ws + alloc((size_t)NNODE * DIN * 2));
  __bf16* disB = (__bf16*)(ws + alloc((size_t)NNODE * DIN * 2));
  __bf16* Wp   = (__bf16*)(ws + alloc((size_t)S * NT * KT1 * 512 * 2));
  float* tmpu  = (float*)(ws + alloc((size_t)S * NNODE * NPAD * 4));
  float* tmpv  = (float*)(ws + alloc((size_t)S * NNODE * NPAD * 4));
  float* hlnc  = (float*)(ws + alloc((size_t)S * NNODE * NPAD * 4));
  float* hdis  = (float*)(ws + alloc((size_t)S * NNODE * NPAD * 4));
  float* g1    = (float*)(ws + alloc(DSUB * 4));
  float* g2    = (float*)(ws + alloc(DSUB * 4));
  float* cc    = (float*)(ws + alloc(2 * 4));
  float* F1l   = (float*)(ws + alloc((size_t)S * NNODE * 4));
  float* F2l   = (float*)(ws + alloc((size_t)S * NNODE * 4));
  float* F1d   = (float*)(ws + alloc((size_t)S * NNODE * 4));
  float* F2d   = (float*)(ws + alloc((size_t)S * NNODE * 4));
  __bf16* Pm   = (__bf16*)(ws + alloc((size_t)NNODE * KPAD * 2));
  float* Zv    = (float*)(ws + alloc((size_t)NNODE * 4));
  __bf16* Hp   = (__bf16*)(ws + alloc((size_t)NT * KT2 * 512 * 2));

  // 1) convert inputs to bf16
  {
    int n = NNODE * DIN;
    cvt_f32_bf16<<<(n + 255) / 256, 256, 0, stream>>>(lnc_in, lncB, n);
    cvt_f32_bf16<<<(n + 255) / 256, 256, 0, stream>>>(dis_in, disB, n);
  }
  // 2) pack weight into WMMA-B layout
  {
    int total = S * NT * KT1 * 512;
    pack_w<<<(total + 255) / 256, 256, 0, stream>>>(weight, Wp);
  }
  // 3) projection GEMMs: tmp_u = lnc @ W[s], tmp_v = dis @ W[s]
  for (int s = 0; s < S; ++s) {
    gemm_bf16<KT1><<<dim3(MTILES, NT / NTW), 32, 0, stream>>>(
        lncB, DIN, Wp + (size_t)s * NT * KT1 * 512, tmpu + (size_t)s * NNODE * NPAD, NPAD);
    gemm_bf16<KT1><<<dim3(MTILES, NT / NTW), 32, 0, stream>>>(
        disB, DIN, Wp + (size_t)s * NT * KT1 * 512, tmpv + (size_t)s * NNODE * NPAD, NPAD);
  }
  // 4) SpMM (zero then atomic scatter).  lnc_h uses tmp_v, dis_h uses tmp_u.
  {
    int n = S * NNODE * NPAD;
    zero_f32<<<(n + 255) / 256, 256, 0, stream>>>(hlnc, n);
    zero_f32<<<(n + 255) / 256, 256, 0, stream>>>(hdis, n);
    spmm_atomic<<<dim3(NNZ, S), 256, 0, stream>>>(lnc_rows, lnc_cols, lnc_vals, tmpv, hlnc);
    spmm_atomic<<<dim3(NNZ, S), 256, 0, stream>>>(dis_rows, dis_cols, dis_vals, tmpu, hdis);
  }
  // 5) fold conv1x1 into g-vectors, then per-node f1/f2
  gvec<<<1, 256, 0, stream>>>(caw, cab, f1w, f1b, f2w, f2b, g1, g2, cc);
  fvals<<<dim3(NNODE, S), 32, 0, stream>>>(hlnc, g1, g2, cc, F1l, F2l);
  fvals<<<dim3(NNODE, S), 32, 0, stream>>>(hdis, g1, g2, cc, F1d, F2d);

  // 6) CRF attention per support-side (buffers reused; stream-ordered)
  for (int side = 0; side < 2; ++side) {
    const float* H  = (side == 0) ? hlnc : hdis;
    const float* Ws = (side == 0) ? w_lnc : w_dis;
    const float* F1 = (side == 0) ? F1l : F1d;
    const float* F2 = (side == 0) ? F2l : F2d;
    float* outBase  = out + (size_t)side * NNODE * S * DSUB;
    for (int s = 0; s < S; ++s) {
      softmax_rows<<<NNODE, 256, 0, stream>>>(F1 + (size_t)s * NNODE, F2 + (size_t)s * NNODE,
                                              Ws, Pm, Zv);
      int total = NT * KT2 * 512;
      pack_h<<<(total + 255) / 256, 256, 0, stream>>>(H + (size_t)s * NNODE * NPAD, Hp);
      attn_gemm<<<dim3(MTILES, NT / NTW), 32, 0, stream>>>(Pm, Hp, Zv,
                                                           H + (size_t)s * NNODE * NPAD, outBase, s);
    }
  }
}
